// MambaGlobalContext_16870631539459
// MI455X (gfx1250) — compile-verified, matrix operand order checked by reading, not run on hardware
//
#include <hip/hip_runtime.h>
#include <hip/hip_bf16.h>

// ---------------------------------------------------------------------------
// Mamba global-context block for MI455X (gfx1250, wave32, WMMA + TDM).
//   B=8, d=256, L=4096, d_state=8, dt_rank=16, d_conv=4
// GEMMs run on v_wmma_f32_16x16x32_bf16 (bf16 in, f32 accum); A-tiles for the
// two large GEMMs are DMA'd into LDS by the Tensor Data Mover with double
// buffering (TENSORcnt), overlapping global->LDS movement with WMMA issue.
// The selective scan is parallelized as a 3-pass chunked linear recurrence.
// ---------------------------------------------------------------------------

#define CH      256
#define L_SEQ   4096
#define BATCH   8
#define DSTATE  8
#define DTRANK  16
#define NCHUNK  64
#define CLEN    (L_SEQ / NCHUNK)   // 64

typedef __bf16 bf16_t;
typedef unsigned int u32;
typedef unsigned long long u64;
typedef __attribute__((ext_vector_type(16))) __bf16 v16bf;
typedef __attribute__((ext_vector_type(8)))  float  v8f;
typedef __attribute__((ext_vector_type(4)))  u32    v4u;
typedef __attribute__((ext_vector_type(8)))  int    v8i;
typedef __attribute__((ext_vector_type(4)))  int    v4i;

// --------------------------- WMMA fragment helpers -------------------------
// 16-bit A/B fragment layout (ISA 7.12.2): lane r = row (M for A, N for B),
// half-wave selects K-octet; VGPR i holds K = 16*(i>=4) + 8*half + 2*(i&3) {+0,+1}.
__device__ __forceinline__ int frag_kbase(int i, int half) {
    return ((i & 4) << 2) | (half << 3) | ((i & 3) << 1);
}

// LDS tile in row-major [row][k] fp32 with row stride `stride` (padded).
__device__ __forceinline__ v16bf frag_from_lds_rowmajor(const float* p, int stride, int lane) {
    const int r = lane & 15, half = (lane >> 4) & 1;
    v16bf f;
#pragma unroll
    for (int i = 0; i < 8; ++i) {
        const int kb = frag_kbase(i, half);
        f[2 * i]     = (bf16_t)p[r * stride + kb];
        f[2 * i + 1] = (bf16_t)p[r * stride + kb + 1];
    }
    return f;
}

// LDS tile in k-major [k][row] fp32 (TDM-transposed tile), row stride `stride`.
__device__ __forceinline__ v16bf frag_from_lds_kmajor(const float* p, int stride, int lane) {
    const int r = lane & 15, half = (lane >> 4) & 1;
    v16bf f;
#pragma unroll
    for (int i = 0; i < 8; ++i) {
        const int kb = frag_kbase(i, half);
        f[2 * i]     = (bf16_t)p[kb * stride + r];
        f[2 * i + 1] = (bf16_t)p[(kb + 1) * stride + r];
    }
    return f;
}

// Row-major fp32 tile in global memory (leading dim ld), converted to bf16.
__device__ __forceinline__ v16bf frag_from_gmem(const float* __restrict__ p, int ld, int lane) {
    const int r = lane & 15, half = (lane >> 4) & 1;
    v16bf f;
#pragma unroll
    for (int i = 0; i < 8; ++i) {
        const int kb = frag_kbase(i, half);
        f[2 * i]     = (bf16_t)p[(size_t)r * ld + kb];
        f[2 * i + 1] = (bf16_t)p[(size_t)r * ld + kb + 1];
    }
    return f;
}

// C/D layout: VGPR j -> M = j + 8*(lane>=16), N = lane & 15.
__device__ __forceinline__ void store_frag(float* __restrict__ dst, int ld, int lane, const v8f& c) {
    const int n = lane & 15, half = (lane >> 4) & 1;
#pragma unroll
    for (int j = 0; j < 8; ++j)
        dst[(size_t)(j + 8 * half) * ld + n] = c[j];
}

__device__ __forceinline__ v8f wmma_bf16(const v16bf& a, const v16bf& b, const v8f& c) {
    return __builtin_amdgcn_wmma_f32_16x16x32_bf16(false, a, false, b, (short)0, c, false, false);
}

__device__ __forceinline__ float silu_f(float v) { return v / (1.f + __expf(-v)); }

// --------------------------- Tensor Data Mover -----------------------------
// Issue a 2D tile DMA global->LDS. D# fields per ISA 8.3/8.4:
//   group0: count=1, lds_addr, global_addr, type=2
//   group1: data_size=4B, pad config, tensor dims/strides, tile dims
// Rows of the tile land consecutively in LDS with 1 DWORD of padding after
// every `1<<(pad_code+1)` DWORDs (bank-conflict avoidance).
// This toolchain exposes the 6-argument builtin:
//   (uint32x4 g0, int32x8 g1, int32x4 g2, int32x4 g3, int32x8 g4, i32 cpol)
__device__ __forceinline__ void tdm_load_2d(u32 lds_off, const float* gaddr,
                                            u32 tensor_d0, u32 tensor_d1,
                                            u32 tile_d0, u32 tile_d1,
                                            u32 stride0, u32 pad_interval_code) {
    const u64 ga = (u64)(uintptr_t)gaddr;
    v4u g0;
    g0[0] = 1u;                                   // count=1, user descriptor
    g0[1] = lds_off;                              // LDS byte address
    g0[2] = (u32)ga;                              // global_addr[31:0]
    g0[3] = (u32)(ga >> 32) | (2u << 30);         // global_addr[56:32] | type=2
    v8i g1;
    g1[0] = (int)((2u << 16)                      // data_size = 4 bytes
                | (1u << 20)                      // pad_enable
                | (pad_interval_code << 22)       // pad every 2^(code+1) DWORDs
                | (0u << 25));                    // pad_amount = 1 DWORD
    g1[1] = (int)(tensor_d0 << 16);               // tensor_dim0[15:0]
    g1[2] = (int)((tensor_d0 >> 16) | (tensor_d1 << 16));
    g1[3] = (int)((tensor_d1 >> 16) | (tile_d0 << 16));
    g1[4] = (int)tile_d1;                         // tile_dim1 (tile_dim2 = 0)
    g1[5] = (int)stride0;                         // tensor_dim0_stride[31:0]
    g1[6] = 0;
    g1[7] = 0;
    const v4i z4 = {0, 0, 0, 0};                  // <=2D: groups 2/3 unused
    const v8i z8 = {0, 0, 0, 0, 0, 0, 0, 0};
    __builtin_amdgcn_tensor_load_to_lds(g0, g1, z4, z4, z8, 0);
}

__device__ __forceinline__ u32 lds_offset_of(const void* p) {
    return (u32)(uintptr_t)p;                     // low 32 bits = LDS byte offset
}

// --------------------------- K1: in_proj GEMM ------------------------------
// xz[(b*L+l), n] = sum_k x[b, k, l] * Win[n, k]   (M=32768, N=512, K=256)
// A is the transpose of x's layout: TDM pulls 16(l) x 32(k) tiles (dim0 = l
// contiguous, 32 rows with stride L) into LDS as [k][l], double buffered.
__global__ void k_gemm_in_proj(const float* __restrict__ x,
                               const float* __restrict__ win,
                               float* __restrict__ xz) {
    __shared__ float At[2][32 * 17];               // k-major, row stride 17
    const int tid = threadIdx.x, lane = tid & 31, wv = tid >> 5;
    const int m0 = blockIdx.x * 16;                // row tile in (b*L)
    const int b  = m0 / L_SEQ;
    const int l0 = m0 % L_SEQ;
    const float* xb = x + (size_t)b * CH * L_SEQ + l0;
    const int n0w = wv * 64;                       // 4 N-tiles per wave
    v8f acc[4] = {};

    if (wv == 0)   // prologue: tile 0   (pad code 3: 16 DWORD rows -> stride 17)
        tdm_load_2d(lds_offset_of(&At[0][0]), xb, L_SEQ, CH, 16, 32, L_SEQ, 3);

    const int NK = CH / 32;                        // 8 K-steps
    for (int kk = 0; kk < NK; ++kk) {
        if (wv == 0) {
            if (kk + 1 < NK) {
                tdm_load_2d(lds_offset_of(&At[(kk + 1) & 1][0]),
                            xb + (size_t)(kk + 1) * 32 * L_SEQ,
                            L_SEQ, CH, 16, 32, L_SEQ, 3);
                __builtin_amdgcn_s_wait_tensorcnt(1);   // tile kk complete
            } else {
                __builtin_amdgcn_s_wait_tensorcnt(0);
            }
        }
        __syncthreads();
        const v16bf af = frag_from_lds_kmajor(&At[kk & 1][0], 17, lane);
        const int k0 = kk * 32;
#pragma unroll
        for (int j = 0; j < 4; ++j) {
            const v16bf bf = frag_from_gmem(win + (size_t)(n0w + 16 * j) * CH + k0, CH, lane);
            acc[j] = wmma_bf16(af, bf, acc[j]);
        }
        __syncthreads();
    }
#pragma unroll
    for (int j = 0; j < 4; ++j)
        store_frag(xz + (size_t)m0 * 512 + (n0w + 16 * j), 512, lane, acc[j]);
}

// --------------------------- K2: causal depthwise conv + silu --------------
__global__ void k_conv_silu(const float* __restrict__ xz,
                            const float* __restrict__ cw,
                            const float* __restrict__ cb,
                            float* __restrict__ xc) {
    const size_t idx = (size_t)blockIdx.x * blockDim.x + threadIdx.x; // b*L*CH
    const int    c   = (int)(idx & (CH - 1));
    const size_t bl  = idx >> 8;                 // b*L + l
    const int    l   = (int)(bl & (L_SEQ - 1));
    float s = cb[c];
#pragma unroll
    for (int j = 0; j < 4; ++j) {
        const int ls = l - 3 + j;
        if (ls >= 0) s += cw[c * 4 + j] * xz[(bl - 3 + j) * 512 + c];
    }
    xc[idx] = silu_f(s);
}

// --------------------------- K3: x_proj GEMM (N=32) ------------------------
__global__ void k_gemm_xdbl(const float* __restrict__ xc,
                            const float* __restrict__ w,
                            float* __restrict__ xdbl) {
    const int tid = threadIdx.x, lane = tid & 31, wv = tid >> 5;
    const int m0 = (blockIdx.x * 8 + wv) * 16;   // one M-tile per wave
    v8f acc0 = {}, acc1 = {};
    for (int k0 = 0; k0 < CH; k0 += 32) {
        const v16bf af = frag_from_gmem(xc + (size_t)m0 * CH + k0, CH, lane);
        const v16bf b0 = frag_from_gmem(w + k0, CH, lane);
        const v16bf b1 = frag_from_gmem(w + (size_t)16 * CH + k0, CH, lane);
        acc0 = wmma_bf16(af, b0, acc0);
        acc1 = wmma_bf16(af, b1, acc1);
    }
    store_frag(xdbl + (size_t)m0 * 32,      32, lane, acc0);
    store_frag(xdbl + (size_t)m0 * 32 + 16, 32, lane, acc1);
}

// --------------------------- K4: dt = softplus(dt_proj) --------------------
__global__ void k_dt(const float* __restrict__ xdbl,
                     const float* __restrict__ dtw,
                     const float* __restrict__ dtb,
                     float* __restrict__ dt) {
    __shared__ float xr[DTRANK];
    const size_t row = blockIdx.x;
    const int d = threadIdx.x;
    if (d < DTRANK) xr[d] = xdbl[row * 32 + d];
    __syncthreads();
    float s = dtb[d];
#pragma unroll
    for (int r = 0; r < DTRANK; ++r) s += xr[r] * dtw[d * DTRANK + r];
    dt[row * CH + d] = (s > 20.f) ? s : __logf(1.f + __expf(s));
}

// --------------------------- K5: chunked selective scan --------------------
// pass1: per (b, chunk, d): a_prod[n] = prod exp(dt*A), h_end[n] with h_in=0
__global__ void k_scan_pass1(const float* __restrict__ dt,
                             const float* __restrict__ xc,
                             const float* __restrict__ xdbl,
                             const float* __restrict__ a_log,
                             float* __restrict__ aprod,
                             float* __restrict__ hend) {
    const int b = blockIdx.x / NCHUNK, chunk = blockIdx.x % NCHUNK;
    const int d = threadIdx.x;
    float A[DSTATE], ap[DSTATE], h[DSTATE];
#pragma unroll
    for (int n = 0; n < DSTATE; ++n) {
        A[n] = -__expf(a_log[d * DSTATE + n]);
        ap[n] = 1.f; h[n] = 0.f;
    }
    const size_t row0 = (size_t)b * L_SEQ + chunk * CLEN;
    for (int t = 0; t < CLEN; ++t) {
        const size_t row = row0 + t;
        const float dtv = dt[row * CH + d];
        const float dx  = dtv * xc[row * CH + d];
        const float* bm = xdbl + row * 32 + DTRANK;
#pragma unroll
        for (int n = 0; n < DSTATE; ++n) {
            const float da = __expf(dtv * A[n]);
            h[n] = da * h[n] + dx * bm[n];
            ap[n] *= da;
        }
    }
    const size_t base = ((size_t)blockIdx.x * CH + d) * DSTATE;
#pragma unroll
    for (int n = 0; n < DSTATE; ++n) { aprod[base + n] = ap[n]; hend[base + n] = h[n]; }
}

// pass2: sequential combine over chunks -> exclusive h_start per chunk
__global__ void k_scan_pass2(const float* __restrict__ aprod,
                             const float* __restrict__ hend,
                             float* __restrict__ hstart) {
    const int idx = blockIdx.x * blockDim.x + threadIdx.x;  // (b*CH + d)*8 + n
    const int b = idx / (CH * DSTATE);
    const int r = idx % (CH * DSTATE);
    const int d = r / DSTATE, n = r % DSTATE;
    float h = 0.f;
    for (int c = 0; c < NCHUNK; ++c) {
        const size_t o = (((size_t)b * NCHUNK + c) * CH + d) * DSTATE + n;
        hstart[o] = h;
        h = aprod[o] * h + hend[o];
    }
}

// pass3: replay with correct h_start; emit u = (y + D*xc) * silu(z)
__global__ void k_scan_pass3(const float* __restrict__ dt,
                             const float* __restrict__ xc,
                             const float* __restrict__ xdbl,
                             const float* __restrict__ xz,
                             const float* __restrict__ a_log,
                             const float* __restrict__ dpar,
                             const float* __restrict__ hstart,
                             float* __restrict__ u) {
    const int b = blockIdx.x / NCHUNK, chunk = blockIdx.x % NCHUNK;
    const int d = threadIdx.x;
    const size_t base = ((size_t)blockIdx.x * CH + d) * DSTATE;
    float A[DSTATE], h[DSTATE];
#pragma unroll
    for (int n = 0; n < DSTATE; ++n) {
        A[n] = -__expf(a_log[d * DSTATE + n]);
        h[n] = hstart[base + n];
    }
    const float Dv = dpar[d];
    const size_t row0 = (size_t)b * L_SEQ + chunk * CLEN;
    for (int t = 0; t < CLEN; ++t) {
        const size_t row = row0 + t;
        const float dtv = dt[row * CH + d];
        const float xv  = xc[row * CH + d];
        const float dx  = dtv * xv;
        const float* bm = xdbl + row * 32 + DTRANK;
        const float* cm = xdbl + row * 32 + DTRANK + DSTATE;
        float y = 0.f;
#pragma unroll
        for (int n = 0; n < DSTATE; ++n) {
            const float da = __expf(dtv * A[n]);
            h[n] = da * h[n] + dx * bm[n];
            y += h[n] * cm[n];
        }
        const float zv = xz[row * 512 + CH + d];
        u[row * CH + d] = (y + xv * Dv) * silu_f(zv);
    }
}

// --------------------------- K6: out_proj GEMM -----------------------------
// yo[(b*L+l), n] = sum_k u[row, k] * Wout[n, k]  (M=32768, N=256, K=256)
// Row-major A: TDM pulls 16(row) x 32(k) tiles (dim0 = k contiguous) into
// LDS as [row][k] with stride 33, double buffered.
__global__ void k_gemm_out(const float* __restrict__ u,
                           const float* __restrict__ w,
                           float* __restrict__ yo) {
    __shared__ float At[2][16 * 33];               // row-major, row stride 33
    const int tid = threadIdx.x, lane = tid & 31, wv = tid >> 5;
    const int m0 = blockIdx.x * 16;
    const int n0w = wv * 32;                       // 2 N-tiles per wave
    const float* ub = u + (size_t)m0 * CH;
    v8f acc[2] = {};

    if (wv == 0)   // prologue: tile 0   (pad code 4: 32 DWORD rows -> stride 33)
        tdm_load_2d(lds_offset_of(&At[0][0]), ub, CH, L_SEQ * BATCH, 32, 16, CH, 4);

    const int NK = CH / 32;                        // 8 K-steps
    for (int kk = 0; kk < NK; ++kk) {
        if (wv == 0) {
            if (kk + 1 < NK) {
                tdm_load_2d(lds_offset_of(&At[(kk + 1) & 1][0]),
                            ub + (size_t)(kk + 1) * 32,
                            CH, L_SEQ * BATCH, 32, 16, CH, 4);
                __builtin_amdgcn_s_wait_tensorcnt(1);
            } else {
                __builtin_amdgcn_s_wait_tensorcnt(0);
            }
        }
        __syncthreads();
        const v16bf af = frag_from_lds_rowmajor(&At[kk & 1][0], 33, lane);
        const int k0 = kk * 32;
#pragma unroll
        for (int j = 0; j < 2; ++j) {
            const v16bf bf = frag_from_gmem(w + (size_t)(n0w + 16 * j) * CH + k0, CH, lane);
            acc[j] = wmma_bf16(af, bf, acc[j]);
        }
        __syncthreads();
    }
#pragma unroll
    for (int j = 0; j < 2; ++j)
        store_frag(yo + (size_t)m0 * CH + (n0w + 16 * j), CH, lane, acc[j]);
}

// --------------------------- K7: LayerNorm + residual (transposing) --------
__global__ void k_ln_res(const float* __restrict__ yo,
                         const float* __restrict__ x,
                         const float* __restrict__ lnw,
                         const float* __restrict__ lnb,
                         float* __restrict__ out) {
    __shared__ float tile[32][CH + 1];
    const int tid = threadIdx.x, lane = tid & 31, wv = tid >> 5;
    const int b  = blockIdx.x / (L_SEQ / 32);
    const int l0 = (blockIdx.x % (L_SEQ / 32)) * 32;
    // each wave normalizes 4 rows (lane-parallel over the 256 channels)
    for (int rr = wv; rr < 32; rr += 8) {
        const size_t row = (size_t)b * L_SEQ + l0 + rr;
        const float* yr = yo + row * CH;
        float v[8], s = 0.f, ss = 0.f;
#pragma unroll
        for (int i = 0; i < 8; ++i) {
            v[i] = yr[lane + 32 * i];
            s += v[i]; ss += v[i] * v[i];
        }
#pragma unroll
        for (int o = 16; o > 0; o >>= 1) {
            s  += __shfl_xor(s,  o, 32);
            ss += __shfl_xor(ss, o, 32);
        }
        const float mu  = s * (1.f / CH);
        const float var = ss * (1.f / CH) - mu * mu;
        const float rs  = rsqrtf(var + 1e-5f);
#pragma unroll
        for (int i = 0; i < 8; ++i) {
            const int c = lane + 32 * i;
            tile[rr][c] = (v[i] - mu) * rs * lnw[c] + lnb[c];
        }
    }
    __syncthreads();
    // transposed, coalesced writeback: out[b, c, l0+l] = x + yn
    const int l = tid & 31, cg = tid >> 5;
#pragma unroll 4
    for (int i = 0; i < 32; ++i) {
        const int c = cg * 32 + i;
        const size_t o = ((size_t)b * CH + c) * L_SEQ + l0 + l;
        out[o] = x[o] + tile[l][c];
    }
}

// --------------------------- launch ----------------------------------------
extern "C" void kernel_launch(void* const* d_in, const int* in_sizes, int n_in,
                              void* d_out, int out_size, void* d_ws, size_t ws_size,
                              hipStream_t stream) {
    (void)in_sizes; (void)n_in; (void)out_size; (void)ws_size;
    const float* x          = (const float*)d_in[0];
    const float* in_proj_w  = (const float*)d_in[1];
    const float* conv_w     = (const float*)d_in[2];
    const float* conv_b     = (const float*)d_in[3];
    const float* x_proj_w   = (const float*)d_in[4];
    const float* dt_proj_w  = (const float*)d_in[5];
    const float* dt_proj_b  = (const float*)d_in[6];
    const float* A_log      = (const float*)d_in[7];
    const float* D_param    = (const float*)d_in[8];
    const float* out_proj_w = (const float*)d_in[9];
    const float* ln_w       = (const float*)d_in[10];
    const float* ln_b       = (const float*)d_in[11];
    float* out = (float*)d_out;
    float* ws  = (float*)d_ws;

    // workspace layout (float offsets)
    const size_t ML = (size_t)BATCH * L_SEQ;               // 32768 rows
    float* xz     = ws;                                    // ML*512
    float* xc     = xz     + ML * 512;                     // ML*256
    float* xdbl   = xc     + ML * CH;                      // ML*32
    float* dtb    = xdbl   + ML * 32;                      // ML*256
    float* ubuf   = dtb    + ML * CH;                      // ML*256
    float* aprod  = ubuf   + ML * CH;                      // B*NCHUNK*CH*8
    float* hend   = aprod  + (size_t)BATCH * NCHUNK * CH * DSTATE;
    float* hstart = hend   + (size_t)BATCH * NCHUNK * CH * DSTATE;
    float* yo     = dtb;   // dt is dead after pass3 -> reuse for out_proj result

    k_gemm_in_proj<<<(int)(ML / 16), 256, 0, stream>>>(x, in_proj_w, xz);
    k_conv_silu  <<<(int)(ML * CH / 256), 256, 0, stream>>>(xz, conv_w, conv_b, xc);
    k_gemm_xdbl  <<<(int)(ML / 128), 256, 0, stream>>>(xc, x_proj_w, xdbl);
    k_dt         <<<(int)ML, 256, 0, stream>>>(xdbl, dt_proj_w, dt_proj_b, dtb);
    k_scan_pass1 <<<BATCH * NCHUNK, 256, 0, stream>>>(dtb, xc, xdbl, A_log, aprod, hend);
    k_scan_pass2 <<<(BATCH * CH * DSTATE) / 256, 256, 0, stream>>>(aprod, hend, hstart);
    k_scan_pass3 <<<BATCH * NCHUNK, 256, 0, stream>>>(dtb, xc, xdbl, xz, A_log, D_param,
                                                      hstart, ubuf);
    k_gemm_out   <<<(int)(ML / 16), 256, 0, stream>>>(ubuf, out_proj_w, yo);
    k_ln_res     <<<BATCH * (L_SEQ / 32), 256, 0, stream>>>(yo, x, ln_w, ln_b, out);
}
